// MultiheadAttention_52063593563035
// MI455X (gfx1250) — matrix operand order repaired and checked by reading, not compiled, not run on hardware
//
#include <hip/hip_runtime.h>
#include <hip/hip_bf16.h>

typedef __attribute__((ext_vector_type(16))) _Float16 v16h;
typedef __attribute__((ext_vector_type(8)))  _Float16 v8h;
typedef __attribute__((ext_vector_type(4)))  _Float16 v4h;
typedef __attribute__((ext_vector_type(8)))  float    v8f;
typedef __attribute__((ext_vector_type(4)))  unsigned int u32x4;
typedef __attribute__((ext_vector_type(8)))  unsigned int u32x8;

#define WMMA_F16(a, b, c) \
  __builtin_amdgcn_wmma_f32_16x16x32_f16(false, (a), false, (b), (short)0, (c), false, false)

// Fast hardware exp2 (v_exp_f32)
#define EXP2F(x) __builtin_amdgcn_exp2f(x)

static constexpr int   Ss = 2048, Ww = 1024, Hh = 16;
static constexpr float ATT_SCALE = 0.3535533905932738f;   // 64^-0.25
static constexpr float L2E       = 1.4426950408889634f;

__device__ __forceinline__ v8f zero8() {
  v8f z;
#pragma unroll
  for (int i = 0; i < 8; ++i) z[i] = 0.0f;
  return z;
}

__device__ __forceinline__ v16h cat16(v8h lo, v8h hi) {
  v16h r;
#pragma unroll
  for (int i = 0; i < 8; ++i) { r[i] = lo[i]; r[8 + i] = hi[i]; }
  return r;
}

// A fragment: 16x32 f16, rows m0..m0+15, cols k0..k0+31, from row-major [., ld]
__device__ __forceinline__ v16h load_a_frag(const _Float16* p, int ld, int m0, int k0, int lane) {
  int m  = m0 + (lane & 15);
  int kb = k0 + ((lane >> 4) << 4);            // +0 or +16
  const _Float16* r = p + m * ld + kb;
  v8h lo = *(const v8h*)(r);                   // K = kb .. kb+7
  v8h hi = *(const v8h*)(r + 8);               // K = kb+8 .. kb+15
  return cat16(lo, hi);
}

// B fragment: 32x16 f16 (KxN); lane loads 16 consecutive K at fixed N (same k order as A).
// Source stored [N][K] row-major with row stride ld.
__device__ __forceinline__ v16h load_b_frag(const _Float16* p, int ld, int n0, int k0, int lane) {
  int n  = n0 + (lane & 15);
  int kb = k0 + ((lane >> 4) << 4);            // +0 or +16
  const _Float16* r = p + n * ld + kb;
  v8h lo = *(const v8h*)(r);
  v8h hi = *(const v8h*)(r + 8);
  return cat16(lo, hi);
}

// TDM: flat 1-D DMA (bytes multiple of 8, bytes/8 <= 65535) global -> LDS.
// D# group0: count=1, lds_addr, 57-bit global byte address, type=2 ("image").
// D# group1: data_size=3 (8B units), tensor_dim0 = tile_dim0 = bytes/8, 1 row.
__device__ __forceinline__ void tdm_load_1d_to_lds(void* lds_ptr, const void* gptr,
                                                   unsigned int bytes) {
  unsigned int lds_off = (unsigned int)(unsigned long long)(uintptr_t)lds_ptr; // addr[31:0] = LDS offset
  unsigned long long ga = (unsigned long long)(uintptr_t)gptr;
  unsigned int n8 = bytes >> 3;

  u32x4 g0;
  g0[0] = 1u;                                           // count=1, is_restore=0, gather=0
  g0[1] = lds_off;                                      // lds_addr
  g0[2] = (unsigned int)(ga & 0xFFFFFFFFu);             // global_addr[31:0]
  g0[3] = (unsigned int)((ga >> 32) & 0x1FFFFFFu)       // global_addr[56:32]
          | (2u << 30);                                 // type=2

  u32x8 g1;
  g1[0] = (3u << 16);                                   // data_size=3 (8 bytes); no pad/iterate
  g1[1] = (n8 & 0xFFFFu) << 16;                         // tensor_dim0[15:0]
  g1[2] = ((n8 >> 16) & 0xFFFFu) | (1u << 16);          // tensor_dim0[31:16], tensor_dim1=1
  g1[3] = (n8 & 0xFFFFu) << 16;                         // tensor_dim1[31:16]=0, tile_dim0 = n8
  g1[4] = 0u;                                           // tile_dim1=0 (unused), tile_dim2=0
  g1[5] = n8;                                           // tensor_dim0_stride[31:0]
  g1[6] = 0u;                                           // stride0[47:32]=0, stride1[15:0]=0
  g1[7] = 0u;

  asm volatile("tensor_load_to_lds %0, %1" :: "s"(g0), "s"(g1) : "memory");
}

__global__ __launch_bounds__(256) void cvt_f32_f16(const float* __restrict__ in,
                                                   _Float16* __restrict__ out, int n) {
  int i = (blockIdx.x * 256 + threadIdx.x) * 4;
  if (i + 3 < n) {
    float4 v = *(const float4*)(in + i);
    v4h o;
    o[0] = (_Float16)v.x; o[1] = (_Float16)v.y; o[2] = (_Float16)v.z; o[3] = (_Float16)v.w;
    *(v4h*)(out + i) = o;
  }
}

// C = X[4096x1024] * Wqkv^T[1024x3072] + b ; epilogue scatters into qh/kh (scaled) and vt (transposed)
// Block = 8 waves sharing one 32-row activation slab (TDM-staged in LDS); waves cover 8 n-tiles.
__global__ __launch_bounds__(256) void qkv_gemm_kernel(const _Float16* __restrict__ x16,
                                                       const _Float16* __restrict__ w16,
                                                       const float* __restrict__ bias,
                                                       _Float16* __restrict__ qh,
                                                       _Float16* __restrict__ kh,
                                                       _Float16* __restrict__ vt) {
  __shared__ _Float16 aslab[32 * 1024];        // 64 KiB activation slab
  const int lane = threadIdx.x & 31;
  const int wave = threadIdx.x >> 5;
  const int gw   = blockIdx.x * 8 + wave;      // [0, 6144)
  const int TN   = (3 * Ww) / 64;              // 48 (multiple of 8 -> block shares tm)
  const int tm = gw / TN, tn = gw % TN;
  const int m0 = tm * 32, n0 = tn * 64;

  if (wave == 0) {
    tdm_load_1d_to_lds(aslab, x16 + (size_t)m0 * Ww, 32u * Ww * 2u);
    __builtin_amdgcn_s_wait_tensorcnt(0);
  }
  __syncthreads();

  v8f acc[2][4];
#pragma unroll
  for (int a = 0; a < 2; ++a)
#pragma unroll
    for (int b = 0; b < 4; ++b) acc[a][b] = zero8();

  for (int k0 = 0; k0 < Ww; k0 += 32) {
    v16h a0 = load_a_frag(aslab, Ww, 0,  k0, lane);    // ds_load from staged slab
    v16h a1 = load_a_frag(aslab, Ww, 16, k0, lane);
#pragma unroll
    for (int j = 0; j < 4; ++j) {
      v16h bf = load_b_frag(w16, Ww, n0 + 16 * j, k0, lane);
      acc[0][j] = WMMA_F16(a0, bf, acc[0][j]);
      acc[1][j] = WMMA_F16(a1, bf, acc[1][j]);
    }
  }

  const int head = n0 / 192;                   // n0 is a multiple of 64
  const int sec  = (n0 / 64) % 3;              // 0=q 1=k 2=v
#pragma unroll
  for (int mi = 0; mi < 2; ++mi)
#pragma unroll
    for (int j = 0; j < 4; ++j)
#pragma unroll
      for (int i = 0; i < 8; ++i) {
        int m = m0 + mi * 16 + i + 8 * (lane >> 4);
        int n = n0 + j * 16 + (lane & 15);
        float val = acc[mi][j][i] + bias[n];
        int bb = m >> 11;                      // / 2048
        int s  = m & 2047;
        int ch = n & 63;
        int base = bb * Hh + head;             // [0, 32)
        if (sec == 0)      qh[(base * Ss + s) * 64 + ch] = (_Float16)(val * ATT_SCALE);
        else if (sec == 1) kh[(base * Ss + s) * 64 + ch] = (_Float16)(val * ATT_SCALE);
        else               vt[(base * 64 + ch) * Ss + s] = (_Float16)val;
      }
}

// Flash attention: one wave per (b*H+h, 32-row q tile). qh/kh: [bh][s][64], vt: [bh][64][s].
// Row sums of P computed with an extra WMMA against an all-ones B fragment.
__global__ __launch_bounds__(256) void attn_kernel(const _Float16* __restrict__ qh,
                                                   const _Float16* __restrict__ kh,
                                                   const _Float16* __restrict__ vt,
                                                   _Float16* __restrict__ o16) {
  __shared__ _Float16 pbuf[8][32 * 32];        // per-wave 32x32 P tile (2 KiB)
  const int lane  = threadIdx.x & 31;
  const int wave  = threadIdx.x >> 5;
  const int gw    = blockIdx.x * 8 + wave;     // [0, 2048)
  const int ttile = gw & 63;                   // 64 tiles of 32 rows per (b,h)
  const int bh    = gw >> 6;                   // [0, 32)
  const _Float16* Q  = qh + bh * Ss * 64;
  const _Float16* Kt = kh + bh * Ss * 64;
  const _Float16* V  = vt + bh * 64 * Ss;
  const int t0 = ttile * 32;

  v16h qf[2][2];                               // [m-frag][ch-chunk]
  qf[0][0] = load_a_frag(Q, 64, t0,      0,  lane);
  qf[0][1] = load_a_frag(Q, 64, t0,      32, lane);
  qf[1][0] = load_a_frag(Q, 64, t0 + 16, 0,  lane);
  qf[1][1] = load_a_frag(Q, 64, t0 + 16, 32, lane);

  v16h onesf;                                  // all-ones 32x16 B matrix
#pragma unroll
  for (int e = 0; e < 16; ++e) onesf[e] = (_Float16)1.0f;

  float mrow[2][8], lrow[2][8];
#pragma unroll
  for (int mi = 0; mi < 2; ++mi)
#pragma unroll
    for (int i = 0; i < 8; ++i) { mrow[mi][i] = -1e30f; lrow[mi][i] = 0.0f; }
  v8f o[2][4];
#pragma unroll
  for (int mi = 0; mi < 2; ++mi)
#pragma unroll
    for (int j = 0; j < 4; ++j) o[mi][j] = zero8();

  _Float16* pb = &pbuf[wave][0];

  for (int s0 = 0; s0 < Ss; s0 += 32) {
    // ---- logits: 32 rows x 32 keys (four 16x16 C tiles) ----
    v16h k00 = load_b_frag(Kt, 64, s0,      0,  lane);
    v16h k01 = load_b_frag(Kt, 64, s0,      32, lane);
    v16h k10 = load_b_frag(Kt, 64, s0 + 16, 0,  lane);
    v16h k11 = load_b_frag(Kt, 64, s0 + 16, 32, lane);
    v8f c[2][2];
#pragma unroll
    for (int mi = 0; mi < 2; ++mi) {
      c[mi][0] = zero8();  c[mi][1] = zero8();
      c[mi][0] = WMMA_F16(qf[mi][0], k00, c[mi][0]);
      c[mi][0] = WMMA_F16(qf[mi][1], k01, c[mi][0]);
      c[mi][1] = WMMA_F16(qf[mi][0], k10, c[mi][1]);
      c[mi][1] = WMMA_F16(qf[mi][1], k11, c[mi][1]);
    }

    // ---- online softmax: running max via shfl tree; P tile -> LDS (f16) ----
    float sc[2][8];
#pragma unroll
    for (int mi = 0; mi < 2; ++mi)
#pragma unroll
      for (int i = 0; i < 8; ++i) {
        float bm = fmaxf(c[mi][0][i], c[mi][1][i]);
#pragma unroll
        for (int msk = 1; msk < 16; msk <<= 1) bm = fmaxf(bm, __shfl_xor(bm, msk, 32));
        float mnew = fmaxf(mrow[mi][i], bm);
        sc[mi][i]  = EXP2F((mrow[mi][i] - mnew) * L2E);
        mrow[mi][i] = mnew;
        float p0 = EXP2F((c[mi][0][i] - mnew) * L2E);
        float p1 = EXP2F((c[mi][1][i] - mnew) * L2E);
        int r = mi * 16 + i + 8 * (lane >> 4);
        pb[r * 32 + (lane & 15)]      = (_Float16)p0;
        pb[r * 32 + 16 + (lane & 15)] = (_Float16)p1;
      }
#pragma unroll
    for (int mi = 0; mi < 2; ++mi)
#pragma unroll
      for (int j = 0; j < 4; ++j)
#pragma unroll
        for (int i = 0; i < 8; ++i) o[mi][j][i] *= sc[mi][i];

    __syncthreads();                           // publish P (C-layout -> A-layout via LDS)
    v16h pf0 = load_a_frag(pb, 32, 0,  0, lane);
    v16h pf1 = load_a_frag(pb, 32, 16, 0, lane);

    // ---- row sums of P via WMMA against ones (replicated per row in C layout) ----
    v8f rs0 = WMMA_F16(pf0, onesf, zero8());
    v8f rs1 = WMMA_F16(pf1, onesf, zero8());
#pragma unroll
    for (int i = 0; i < 8; ++i) {
      lrow[0][i] = lrow[0][i] * sc[0][i] + rs0[i];
      lrow[1][i] = lrow[1][i] * sc[1][i] + rs1[i];
    }

    // ---- PV: o += P(32x32) x V(32x64) ----
#pragma unroll
    for (int j = 0; j < 4; ++j) {
      v16h vf = load_b_frag(V, Ss, j * 16, s0, lane);   // N=ch, K=s (vt is [ch][s])
      o[0][j] = WMMA_F16(pf0, vf, o[0][j]);
      o[1][j] = WMMA_F16(pf1, vf, o[1][j]);
    }
    __syncthreads();
  }

  const int bb = bh >> 4, h = bh & 15;
#pragma unroll
  for (int mi = 0; mi < 2; ++mi)
#pragma unroll
    for (int j = 0; j < 4; ++j)
#pragma unroll
      for (int i = 0; i < 8; ++i) {
        int t  = t0 + mi * 16 + i + 8 * (lane >> 4);
        int ch = j * 16 + (lane & 15);
        float val = o[mi][j][i] / lrow[mi][i];
        o16[(bb * Ss + t) * Ww + h * 64 + ch] = (_Float16)val;
      }
}

// out = O[4096x1024] * Wproj^T[1024x1024] + b_proj (f32 output); TDM-staged A slab.
__global__ __launch_bounds__(256) void proj_gemm_kernel(const _Float16* __restrict__ o16,
                                                        const _Float16* __restrict__ w16,
                                                        const float* __restrict__ bias,
                                                        float* __restrict__ out) {
  __shared__ _Float16 aslab[32 * 1024];        // 64 KiB
  const int lane = threadIdx.x & 31;
  const int wave = threadIdx.x >> 5;
  const int gw   = blockIdx.x * 8 + wave;      // [0, 2048)
  const int tm = gw >> 4, tn = gw & 15;        // TN=16, multiple of 8 -> block shares tm
  const int m0 = tm * 32, n0 = tn * 64;

  if (wave == 0) {
    tdm_load_1d_to_lds(aslab, o16 + (size_t)m0 * Ww, 32u * Ww * 2u);
    __builtin_amdgcn_s_wait_tensorcnt(0);
  }
  __syncthreads();

  v8f acc[2][4];
#pragma unroll
  for (int a = 0; a < 2; ++a)
#pragma unroll
    for (int b = 0; b < 4; ++b) acc[a][b] = zero8();

  for (int k0 = 0; k0 < Ww; k0 += 32) {
    v16h a0 = load_a_frag(aslab, Ww, 0,  k0, lane);
    v16h a1 = load_a_frag(aslab, Ww, 16, k0, lane);
#pragma unroll
    for (int j = 0; j < 4; ++j) {
      v16h bf = load_b_frag(w16, Ww, n0 + 16 * j, k0, lane);
      acc[0][j] = WMMA_F16(a0, bf, acc[0][j]);
      acc[1][j] = WMMA_F16(a1, bf, acc[1][j]);
    }
  }

#pragma unroll
  for (int mi = 0; mi < 2; ++mi)
#pragma unroll
    for (int j = 0; j < 4; ++j)
#pragma unroll
      for (int i = 0; i < 8; ++i) {
        int m = m0 + mi * 16 + i + 8 * (lane >> 4);
        int n = n0 + j * 16 + (lane & 15);
        out[m * Ww + n] = acc[mi][j][i] + bias[n];
      }
}

extern "C" void kernel_launch(void* const* d_in, const int* in_sizes, int n_in,
                              void* d_out, int out_size, void* d_ws, size_t ws_size,
                              hipStream_t stream) {
  const float* x      = (const float*)d_in[0];   // [2,2048,1024]
  const float* w_qkv  = (const float*)d_in[1];   // [3072,1024]
  const float* b_qkv  = (const float*)d_in[2];   // [3072]
  const float* w_proj = (const float*)d_in[3];   // [1024,1024]
  const float* b_proj = (const float*)d_in[4];   // [1024]
  float* out = (float*)d_out;

  char* ws = (char*)d_ws;
  _Float16* x16     = (_Float16*)(ws);                        //  8 MiB
  _Float16* wqkv16  = (_Float16*)(ws + (size_t)8  * 1048576); //  6 MiB
  _Float16* wproj16 = (_Float16*)(ws + (size_t)14 * 1048576); //  2 MiB
  _Float16* qh      = (_Float16*)(ws + (size_t)16 * 1048576); //  8 MiB  [32][2048][64]
  _Float16* kh      = (_Float16*)(ws + (size_t)24 * 1048576); //  8 MiB  [32][2048][64]
  _Float16* vt      = (_Float16*)(ws + (size_t)32 * 1048576); //  8 MiB  [32][64][2048]
  _Float16* o16     = (_Float16*)(ws + (size_t)40 * 1048576); //  8 MiB  [4096][1024]

  cvt_f32_f16<<<4096, 256, 0, stream>>>(x,      x16,     4194304);
  cvt_f32_f16<<<3072, 256, 0, stream>>>(w_qkv,  wqkv16,  3145728);
  cvt_f32_f16<<<1024, 256, 0, stream>>>(w_proj, wproj16, 1048576);

  qkv_gemm_kernel<<<768, 256, 0, stream>>>(x16, wqkv16, b_qkv, qh, kh, vt);
  attn_kernel<<<256, 256, 0, stream>>>(qh, kh, vt, o16);
  proj_gemm_kernel<<<256, 256, 0, stream>>>(o16, wproj16, b_proj, out);
}
